// QLinear_75161927680363
// MI455X (gfx1250) — matrix operand order repaired and checked by reading, not compiled, not run on hardware
//
#include <hip/hip_runtime.h>
#include <stdint.h>

#define K_DIM 4096
#define N_DIM 11008
#define GS 32
#define M_TOKENS 32
#define K_SPLIT 8
#define K_SLICE (K_DIM / K_SPLIT)   // 512 -> 16 WMMA steps per wave

typedef __attribute__((ext_vector_type(16))) __bf16   v16bf;
typedef __attribute__((ext_vector_type(8)))  float    v8f;
typedef __attribute__((ext_vector_type(4)))  int      v4i;
typedef __attribute__((ext_vector_type(4)))  uint32_t v4u;

union BF16x16 {
    v16bf    v;
    uint32_t u[8];
};

#define USE_ASM_CVT_PK_BF16 1

// Pack two f32 into one dword of two bf16 (lo = a, hi = b).
__device__ __forceinline__ uint32_t cvt_pk_bf16(float a, float b) {
#if USE_ASM_CVT_PK_BF16
    uint32_t r;
    asm("v_cvt_pk_bf16_f32 %0, %1, %2" : "=v"(r) : "v"(a), "v"(b));
    return r;
#else
    uint32_t ua = __builtin_bit_cast(uint32_t, a);
    uint32_t ub = __builtin_bit_cast(uint32_t, b);
    ua = (ua + 0x7FFFu + ((ua >> 16) & 1u)) >> 16;          // low half (RNE)
    ub = (ub + 0x7FFFu + ((ub >> 16) & 1u)) & 0xFFFF0000u;  // high half (RNE)
    return ub | ua;
#endif
}

// ---- Prep 1: x[32,4096] f32 -> packed bf16 in ws (one dword = 2 bf16) ----
__global__ __launch_bounds__(256)
void x_to_bf16(const float* __restrict__ x, uint32_t* __restrict__ xbf) {
    const int i = blockIdx.x * blockDim.x + threadIdx.x;   // 0 .. 65535
    const float2 p = ((const float2*)x)[i];
    xbf[i] = cvt_pk_bf16(p.x, p.y);
}

// ---- Prep 2: y[m,n] = bias[n] (split-K waves then atomically accumulate) ----
__global__ __launch_bounds__(256)
void y_init_bias(const float* __restrict__ bias, float* __restrict__ y) {
    const int n = blockIdx.x * blockDim.x + threadIdx.x;   // 0 .. 11007
    const int m = blockIdx.y;                              // 0 .. 31
    y[(size_t)m * N_DIM + n] = bias[n];
}

// ---- Main: W4 group-dequant GEMM slice via v_wmma_f32_16x16x32_bf16 ----
__global__ __launch_bounds__(32)
void qlinear_wmma_bf16(const uint32_t* __restrict__ xbf,  // [32, 4096] bf16 packed
                       const int*      __restrict__ wq,   // [11008, 4096] int4-in-int32
                       const float*    __restrict__ sz,   // [128, 11008, 2] (scale, zero)
                       float*          __restrict__ y)    // [32, 11008], pre-set to bias
{
    const int lane  = threadIdx.x;          // wave32
    const int n0    = blockIdx.x * 16;      // 16 output features per wave
    const int kBase = blockIdx.y * K_SLICE; // this wave's K slice
    const int half  = lane >> 4;
    const int l16   = lane & 15;

    const int col = n0 + l16;               // output feature owned by this lane

    // A-fragment K striping (16-bit A 16x32): regs 0..3 <- K aoff0..aoff0+7
    // (16 contiguous bytes), regs 4..7 <- K aoff1..aoff1+7.
    const int adw0 = (half ? 16 : 0)  >> 1;   // dword offset in K-chunk
    const int adw1 = (half ? 24 : 8) >> 1;
    // B-fragment: lane = N column; low lanes K 0..15, high lanes K 16..31.
    const int boff = half ? 16 : 0;

    const uint32_t* xr0 = xbf + (size_t)l16        * (K_DIM / 2); // M tile 0 row
    const uint32_t* xr1 = xbf + (size_t)(16 + l16) * (K_DIM / 2); // M tile 1 row
    const int*      wN  = wq  + (size_t)col * K_DIM + boff;

    v8f acc0 = {};   // M rows 0..15
    v8f acc1 = {};   // M rows 16..31

#pragma unroll 2
    for (int k0 = kBase; k0 < kBase + K_SLICE; k0 += GS) {
        const int g  = k0 >> 5;   // quant group == WMMA K chunk
        const int kd = k0 >> 1;   // packed-bf16 dword base

        // Per-lane scale/zero (b64 load); fold midpoint: w = q*s + (z - 8s)
        const float2 p = *(const float2*)(sz + ((size_t)g * N_DIM + col) * 2);
        const float s = p.x, z = p.y - 8.0f * p.x;

        // ---- A fragments: pre-converted bf16, 2x b128 per fragment ----
        const v4u a0lo = *(const v4u*)(xr0 + kd + adw0);
        const v4u a0hi = *(const v4u*)(xr0 + kd + adw1);
        const v4u a1lo = *(const v4u*)(xr1 + kd + adw0);
        const v4u a1hi = *(const v4u*)(xr1 + kd + adw1);
        BF16x16 a0, a1;
#pragma unroll
        for (int r = 0; r < 4; ++r) {
            a0.u[r]     = a0lo[r];
            a0.u[4 + r] = a0hi[r];
            a1.u[r]     = a1lo[r];
            a1.u[4 + r] = a1hi[r];
        }

        // ---- B fragment: 16 consecutive int32 per lane via 4x b128 NT loads ----
        BF16x16 b;
#pragma unroll
        for (int r4 = 0; r4 < 4; ++r4) {
            const v4i q = __builtin_nontemporal_load((const v4i*)(wN + k0) + r4);
            b.u[2 * r4 + 0] = cvt_pk_bf16((float)q[0] * s + z, (float)q[1] * s + z);
            b.u[2 * r4 + 1] = cvt_pk_bf16((float)q[2] * s + z, (float)q[3] * s + z);
        }

        // Prefetch the weight stream ~1KB (8 steps) ahead into GL2.
        if (k0 + 256 < K_DIM) {
            __builtin_prefetch(wN + k0 + 256, 0, 1);
        }

        // ---- 2 WMMAs: 2 M tiles x 1 N tile, B shared across M ----
        acc0 = __builtin_amdgcn_wmma_f32_16x16x32_bf16(false, a0.v, false, b.v,
                                                       (short)0, acc0, false, false);
        acc1 = __builtin_amdgcn_wmma_f32_16x16x32_bf16(false, a1.v, false, b.v,
                                                       (short)0, acc1, false, false);
    }

    // ---- Epilogue: C/D layout (lane = N, VGPR r -> row r / 8+r) ----
    // Split-K partial sums accumulate with native f32 atomics.
#pragma unroll
    for (int r = 0; r < 8; ++r) {
        const int m = (half ? 8 : 0) + r;
        unsafeAtomicAdd(&y[(size_t)m        * N_DIM + col], acc0[r]);
        unsafeAtomicAdd(&y[(size_t)(16 + m) * N_DIM + col], acc1[r]);
    }
}

extern "C" void kernel_launch(void* const* d_in, const int* in_sizes, int n_in,
                              void* d_out, int out_size, void* d_ws, size_t ws_size,
                              hipStream_t stream) {
    (void)in_sizes; (void)n_in; (void)out_size; (void)ws_size;
    const float* x    = (const float*)d_in[0];  // [32, 4096]
    const int*   wq   = (const int*)d_in[1];    // [11008, 4096]
    const float* sz   = (const float*)d_in[2];  // [128, 11008, 2]
    const float* bias = (const float*)d_in[3];  // [11008]
    float*       y    = (float*)d_out;          // [32, 11008]
    uint32_t*    xbf  = (uint32_t*)d_ws;        // 256 KB packed-bf16 activations

    // 1) Convert activations to bf16 once (L2-resident afterwards).
    x_to_bf16<<<dim3((M_TOKENS * K_DIM / 2) / 256), dim3(256), 0, stream>>>(x, xbf);
    // 2) y = bias (atomic split-K accumulation target).
    y_init_bias<<<dim3(N_DIM / 256, M_TOKENS), dim3(256), 0, stream>>>(bias, y);
    // 3) Streaming dequant GEMM: 688 N-tiles x 8 K-slices = 5504 waves.
    qlinear_wmma_bf16<<<dim3(N_DIM / 16, K_SPLIT), dim3(32), 0, stream>>>(xbf, wq, sz, y);
}
